// PredCodingMultipleChoice_35416300323648
// MI455X (gfx1250) — matrix operand order repaired and checked by reading, compiled, not run on hardware
//
#include <hip/hip_runtime.h>

// ---------------------------------------------------------------------------
// PredCodingMultipleChoice on MI455X (gfx1250, wave32, WMMA)
//
// B=1024 C=4 S=512 D=64 V=32000 L=4 ITERS=10  ->  N = B*C = 4096 rows
//
// Phase 1 (bandwidth-bound): embedding gather-pool, 512MB of gathers from an
//   8MB (L2-resident) table. Wave-per-row, coalesced 256B row fetches.
// Phase 2 (fused, LDS-resident): all 12 D x D weight mats (bf16, 96KB) + all
//   activations for a 64-row tile live in LDS; init pass + 10 PC iterations +
//   scorer run with zero global traffic. GEMMs use v_wmma_f32_16x16x32_bf16
//   (f32 accumulate); reps stay f32 in LDS across iterations.
//
// Round-1 fixes: branchless tanh-GELU (native v_exp + v_rcp, no EXEC
// divergence in the hot epilogue), float4/b128 LDS traffic for A-packs and
// elementwise passes.
// ---------------------------------------------------------------------------

typedef __attribute__((ext_vector_type(16))) __bf16 v16bf;
typedef __attribute__((ext_vector_type(8)))  float  v8f;

#define D_      64
#define S_      512
#define N_      4096
#define L_      4
#define ITERS_  10
#define LR_     0.1f

#define WAVES_  4
#define ROWS_   (WAVES_ * 16)        // rows per block in pc_kernel
#define RD_     (ROWS_ * D_)         // floats per activation buffer

// LDS budget for pc_kernel (must match the carve below):
//   bf16 weights: (3*L*D*D + D*D/2)*2      = 102400 B
//   f32  biases : (3*L*D + 32 + 32)*4      =   3328 B
//   f32  acts   : (9*RD + ROWS*32)*4       = 155648 B
static constexpr size_t kPcLds =
    (size_t)(3 * L_ * D_ * D_ + D_ * (D_ / 2)) * 2 +
    (size_t)(3 * L_ * D_ + (D_ / 2) + (D_ / 2)) * 4 +
    (size_t)(9 * RD_ + ROWS_ * 32) * 4;            // = 261376 B  (< 320KB WGP)

// Branchless GELU (tanh form). tanh(u) = (1-e^-2u)/(1+e^-2u) with native
// v_exp_f32; u clamped so e^-2u stays finite -> no NaN, no branches, and the
// TRANS ops co-execute with neighboring WMMA/VALU.
__device__ __forceinline__ float gelu_f(float x)
{
    float u = 0.7978845608028654f * x * (1.0f + 0.044715f * x * x);
    u = fminf(fmaxf(u, -15.0f), 15.0f);
    const float e2 = __expf(-2.0f * u);                       // v_exp_f32
    const float th = (1.0f - e2) * __builtin_amdgcn_rcpf(1.0f + e2); // v_rcp
    return 0.5f * x * (1.0f + th);
}

// ---------------------------------------------------------------------------
// Kernel 0: pos-mean + fp32 -> bf16 weight conversion
// ---------------------------------------------------------------------------
__global__ void __launch_bounds__(256)
prep_kernel(const float* __restrict__ pos,
            const float* __restrict__ iW, const float* __restrict__ pW,
            const float* __restrict__ uW, const float* __restrict__ s1,
            float* __restrict__ posmean,
            __bf16* __restrict__ bWI, __bf16* __restrict__ bWP,
            __bf16* __restrict__ bWU, __bf16* __restrict__ bW1)
{
    const int tid = blockIdx.x * blockDim.x + threadIdx.x;
    const int nth = gridDim.x * blockDim.x;

    if (blockIdx.x == 0 && threadIdx.x < D_) {
        float s = 0.0f;
        for (int t = 0; t < S_; ++t) s += pos[t * D_ + threadIdx.x];
        posmean[threadIdx.x] = s * (1.0f / (float)S_);
    }
    for (int i = tid; i < L_ * D_ * D_; i += nth) {
        bWI[i] = (__bf16)iW[i];
        bWP[i] = (__bf16)pW[i];
        bWU[i] = (__bf16)uW[i];
    }
    for (int i = tid; i < D_ * (D_ / 2); i += nth) bW1[i] = (__bf16)s1[i];
}

// ---------------------------------------------------------------------------
// Kernel 1: embedding gather-pool.  One wave per row; 32 ids loaded coalesced,
// broadcast by shfl; each lane owns a float2 slice of the 256B embedding row.
// ---------------------------------------------------------------------------
__global__ void __launch_bounds__(256)
pool_kernel(const int* __restrict__ ids, const float* __restrict__ emb,
            const float* __restrict__ posmean, float* __restrict__ pooled)
{
    const int lane = threadIdx.x & 31;
    const int wave = threadIdx.x >> 5;
    const int row  = blockIdx.x * 8 + wave;          // 4096/8 = 512 blocks
    const int* idrow = ids + (size_t)row * S_;

    float ax = 0.0f, ay = 0.0f;
    for (int s0 = 0; s0 < S_; s0 += 32) {
        const int myid = idrow[s0 + lane];           // coalesced 128B id fetch
        if (s0 + 32 < S_)
            __builtin_prefetch(idrow + s0 + 32 + lane, 0, 0); // global_prefetch_b8
#pragma unroll 8
        for (int j = 0; j < 32; ++j) {
            const int id = __shfl(myid, j, 32);
            const float2 v =
                *(const float2*)(emb + (size_t)id * D_ + lane * 2);
            ax += v.x; ay += v.y;
        }
    }
    const float inv = 1.0f / (float)S_;
    float2 o;
    o.x = ax * inv + posmean[lane * 2 + 0];
    o.y = ay * inv + posmean[lane * 2 + 1];
    *(float2*)(pooled + (size_t)row * D_ + lane * 2) = o;
}

// ---------------------------------------------------------------------------
// Wave-level GEMM tile: 16 rows x ncols = [16x64] @ [64 x ncols] + bias.
//   in   : f32 LDS, row stride D_ (rows rowbase..rowbase+15 of this wave)
//   w    : bf16 LDS, row-major [64 in][ncols out]
//   out  : f32 LDS, row stride ncols
// A-pack per the CDNA5 16-bit 16x32 layout: lane = half*16+m, VGPR e holds
// K = {half*8+e, 16+half*8+e} (+32 for the second K-tile). Loaded as
// float4 -> ds_load_b128 (all chunks 32B aligned by construction).
// B-pack: lane holds column n = nt*16 + mlo, 16 bf16 over K in the A-pack order (+32 for b1).
// MODE: 0 = store linear, 1 = store GELU, 2 = out += scale*val
// ---------------------------------------------------------------------------
template <int MODE>
__device__ __forceinline__ void wave_gemm(const float* __restrict__ in,
                                          const __bf16* __restrict__ w,
                                          const float* __restrict__ bias,
                                          float* __restrict__ out,
                                          int ncols, int rowbase, int lane,
                                          float scale)
{
    const int mlo  = lane & 15;
    const int half = lane >> 4;

    v16bf a0, a1;
    {
        const float4* p =
            (const float4*)(in + (rowbase + mlo) * D_ + half * 8);
        const float4 c0 = p[0],  c1 = p[1];    // K =      half*8 + 0..7
        const float4 c2 = p[4],  c3 = p[5];    // K = 16 + half*8 + 0..7
        const float4 c4 = p[8],  c5 = p[9];    // K = 32 + half*8 + 0..7
        const float4 c6 = p[12], c7 = p[13];   // K = 48 + half*8 + 0..7
        a0[0]  = (__bf16)c0.x; a0[1]  = (__bf16)c0.y;
        a0[2]  = (__bf16)c0.z; a0[3]  = (__bf16)c0.w;
        a0[4]  = (__bf16)c1.x; a0[5]  = (__bf16)c1.y;
        a0[6]  = (__bf16)c1.z; a0[7]  = (__bf16)c1.w;
        a0[8]  = (__bf16)c2.x; a0[9]  = (__bf16)c2.y;
        a0[10] = (__bf16)c2.z; a0[11] = (__bf16)c2.w;
        a0[12] = (__bf16)c3.x; a0[13] = (__bf16)c3.y;
        a0[14] = (__bf16)c3.z; a0[15] = (__bf16)c3.w;
        a1[0]  = (__bf16)c4.x; a1[1]  = (__bf16)c4.y;
        a1[2]  = (__bf16)c4.z; a1[3]  = (__bf16)c4.w;
        a1[4]  = (__bf16)c5.x; a1[5]  = (__bf16)c5.y;
        a1[6]  = (__bf16)c5.z; a1[7]  = (__bf16)c5.w;
        a1[8]  = (__bf16)c6.x; a1[9]  = (__bf16)c6.y;
        a1[10] = (__bf16)c6.z; a1[11] = (__bf16)c6.w;
        a1[12] = (__bf16)c7.x; a1[13] = (__bf16)c7.y;
        a1[14] = (__bf16)c7.z; a1[15] = (__bf16)c7.w;
    }

    const int ntiles = ncols >> 4;
    for (int nt = 0; nt < ntiles; ++nt) {
        v8f c;
#pragma unroll
        for (int e = 0; e < 8; ++e) c[e] = 0.0f;

        v16bf b0, b1;
#pragma unroll
        for (int s = 0; s < 16; ++s) {
            const int k = (s < 8) ? (half * 8 + s) : (16 + half * 8 + (s - 8));
            b0[s] = w[k * ncols + nt * 16 + mlo];
            b1[s] = w[(32 + k) * ncols + nt * 16 + mlo];
        }

        c = __builtin_amdgcn_wmma_f32_16x16x32_bf16(false, a0, false, b0,
                                                    (short)0, c, false, false);
        c = __builtin_amdgcn_wmma_f32_16x16x32_bf16(false, a1, false, b1,
                                                    (short)0, c, false, false);

        const float bv = bias[nt * 16 + mlo];    // bias depends on column only
#pragma unroll
        for (int r = 0; r < 8; ++r) {            // element (m = r + half*8, n = mlo)
            float v = c[r] + bv;
            if (MODE == 1) v = gelu_f(v);        // branchless, TRANS-based
            float* dst = out + (rowbase + r + half * 8) * ncols + nt * 16 + mlo;
            if (MODE == 2) *dst += scale * v;
            else           *dst = v;
        }
    }
}

// ---------------------------------------------------------------------------
// Kernel 2: fused PC network. 64 rows/block, each wave owns 16 rows end-to-end
// (rows are independent -> single __syncthreads after cooperative staging).
// ---------------------------------------------------------------------------
__global__ void __launch_bounds__(WAVES_ * 32)
pc_kernel(const float* __restrict__ pooled,
          const __bf16* __restrict__ gWI, const __bf16* __restrict__ gWP,
          const __bf16* __restrict__ gWU, const __bf16* __restrict__ gW1,
          const float* __restrict__ gBI, const float* __restrict__ gBP,
          const float* __restrict__ gBU, const float* __restrict__ gB1,
          const float* __restrict__ gW2, const float* __restrict__ gB2,
          float* __restrict__ out)
{
    extern __shared__ __attribute__((aligned(128))) unsigned char smem_raw[];
    __bf16* sWI = (__bf16*)smem_raw;                 // L*D*D
    __bf16* sWP = sWI + L_ * D_ * D_;
    __bf16* sWU = sWP + L_ * D_ * D_;
    __bf16* sW1 = sWU + L_ * D_ * D_;                // D * D/2
    float*  sBI = (float*)(sW1 + D_ * (D_ / 2));     // byte off 102400 (32B ok)
    float*  sBP = sBI + L_ * D_;
    float*  sBU = sBP + L_ * D_;
    float*  sB1 = sBU + L_ * D_;                     // 32
    float*  sW2 = sB1 + (D_ / 2);                    // 32
    float*  sPool = sW2 + (D_ / 2);                  // RD_
    float*  sReps = sPool + RD_;                     // 4 * RD_
    float*  sFF   = sReps + L_ * RD_;                // RD_
    float*  sPc   = sFF + RD_;                       // RD_
    float*  sPp   = sPc + RD_;                       // RD_
    float*  sErr  = sPp + RD_;                       // RD_
    float*  sHid  = sErr + RD_;                      // ROWS_*32

    const int tid     = threadIdx.x;
    const int lane    = tid & 31;
    const int wave    = tid >> 5;
    const int rowbase = wave * 16;
    const int grow    = blockIdx.x * ROWS_;

    // ---- cooperative staging (only cross-wave dependency in the kernel) ----
    for (int i = tid; i < L_ * D_ * D_; i += blockDim.x) {
        sWI[i] = gWI[i]; sWP[i] = gWP[i]; sWU[i] = gWU[i];
    }
    for (int i = tid; i < D_ * (D_ / 2); i += blockDim.x) sW1[i] = gW1[i];
    for (int i = tid; i < L_ * D_; i += blockDim.x) {
        sBI[i] = gBI[i]; sBP[i] = gBP[i]; sBU[i] = gBU[i];
    }
    if (tid < D_ / 2) { sB1[tid] = gB1[tid]; sW2[tid] = gW2[tid]; }
    for (int i = tid; i < RD_; i += blockDim.x)
        sPool[i] = pooled[(size_t)grow * D_ + i];
    __syncthreads();

    // ---- feed-forward init pass: cur = gelu(cur @ initW[l] + b) ----
    {
        const float* cur = sPool;
        for (int l = 0; l < L_; ++l) {
            wave_gemm<1>(cur, sWI + l * D_ * D_, sBI + l * D_,
                         sReps + l * RD_, D_, rowbase, lane, 0.0f);
            cur = sReps + l * RD_;
        }
    }
    // ff_output = reps[3]  (wave-private rows, float4 / ds_b128)
    {
        const float4* src = (const float4*)(sReps + 3 * RD_);
        float4*       dst = (float4*)sFF;
        for (int q = lane; q < 16 * (D_ / 4); q += 32) {
            const int idx = (rowbase + (q >> 4)) * (D_ / 4) + (q & 15);
            dst[idx] = src[idx];
        }
    }

    // ---- 10 PC refinement iterations; process layers top-down so only
    //      preds[i] (pcb) and preds[i+1] (ppb) are live ----
    float* pcb = sPc;
    float* ppb = sPp;
    for (int it = 0; it < ITERS_; ++it) {
        for (int i = L_ - 1; i >= 0; --i) {
            // preds[i] = reps[i] @ predW[i] + predb[i]
            wave_gemm<0>(sReps + i * RD_, sWP + i * D_ * D_, sBP + i * D_,
                         pcb, D_, rowbase, lane, 0.0f);
            // total_err = (below - preds[i]) + 0.5*(reps[i] - preds[i+1])
            const float* below = (i == 0) ? sPool : (sReps + (i - 1) * RD_);
            {
                const float4* b4 = (const float4*)below;
                const float4* p4 = (const float4*)pcb;
                const float4* r4 = (const float4*)(sReps + i * RD_);
                const float4* q4 = (const float4*)ppb;
                float4*       e4 = (float4*)sErr;
                for (int q = lane; q < 16 * (D_ / 4); q += 32) {
                    const int idx = (rowbase + (q >> 4)) * (D_ / 4) + (q & 15);
                    const float4 bb = b4[idx], pp = p4[idx];
                    float4 t;
                    t.x = bb.x - pp.x; t.y = bb.y - pp.y;
                    t.z = bb.z - pp.z; t.w = bb.w - pp.w;
                    if (i < L_ - 1) {
                        const float4 rr = r4[idx], qq = q4[idx];
                        t.x += 0.5f * (rr.x - qq.x);
                        t.y += 0.5f * (rr.y - qq.y);
                        t.z += 0.5f * (rr.z - qq.z);
                        t.w += 0.5f * (rr.w - qq.w);
                    }
                    e4[idx] = t;
                }
            }
            // reps[i] += LR * (total_err @ updW[i] + updb[i])
            wave_gemm<2>(sErr, sWU + i * D_ * D_, sBU + i * D_,
                         sReps + i * RD_, D_, rowbase, lane, LR_);
            float* t = ppb; ppb = pcb; pcb = t;   // ppb <- preds[i]
        }
    }

    // ---- rep = reps[3] + ff_output ----
    {
        float4*       r4 = (float4*)(sReps + 3 * RD_);
        const float4* f4 = (const float4*)sFF;
        for (int q = lane; q < 16 * (D_ / 4); q += 32) {
            const int idx = (rowbase + (q >> 4)) * (D_ / 4) + (q & 15);
            float4 a = r4[idx];
            const float4 b = f4[idx];
            a.x += b.x; a.y += b.y; a.z += b.z; a.w += b.w;
            r4[idx] = a;
        }
    }

    // ---- scorer: gelu(rep @ W1 + b1) @ W2 + b2 ----
    wave_gemm<1>(sReps + 3 * RD_, sW1, sB1, sHid, 32, rowbase, lane, 0.0f);

    if (lane < 16) {                                 // after all WMMAs (EXEC ok)
        const int row = rowbase + lane;
        float s = gB2[0];
#pragma unroll
        for (int j = 0; j < 32; ++j) s += sHid[row * 32 + j] * sW2[j];
        out[grow + row] = s;
    }
}

// ---------------------------------------------------------------------------
extern "C" void kernel_launch(void* const* d_in, const int* in_sizes, int n_in,
                              void* d_out, int out_size, void* d_ws,
                              size_t ws_size, hipStream_t stream)
{
    (void)in_sizes; (void)n_in; (void)out_size; (void)ws_size;

    const int*   ids   = (const int*)  d_in[0];
    const float* emb   = (const float*)d_in[1];
    const float* pos   = (const float*)d_in[2];
    const float* initW = (const float*)d_in[3];
    const float* initB = (const float*)d_in[4];
    const float* predW = (const float*)d_in[5];
    const float* predB = (const float*)d_in[6];
    const float* updW  = (const float*)d_in[7];
    const float* updB  = (const float*)d_in[8];
    const float* s1W   = (const float*)d_in[9];
    const float* s1B   = (const float*)d_in[10];
    const float* s2W   = (const float*)d_in[11];
    const float* s2B   = (const float*)d_in[12];
    float* out = (float*)d_out;

    // workspace carve (all segments 256B-aligned)
    unsigned char* ws = (unsigned char*)d_ws;
    float*  posmean = (float*)ws;  ws += 256;
    __bf16* bWI = (__bf16*)ws;     ws += (size_t)L_ * D_ * D_ * 2;   // 32KB
    __bf16* bWP = (__bf16*)ws;     ws += (size_t)L_ * D_ * D_ * 2;
    __bf16* bWU = (__bf16*)ws;     ws += (size_t)L_ * D_ * D_ * 2;
    __bf16* bW1 = (__bf16*)ws;     ws += (size_t)D_ * (D_ / 2) * 2 + 256;
    float*  pooled = (float*)ws;   ws += (size_t)N_ * D_ * 4;        // 1MB

    prep_kernel<<<64, 256, 0, stream>>>(pos, initW, predW, updW, s1W,
                                        posmean, bWI, bWP, bWU, bW1);

    pool_kernel<<<N_ / 8, 256, 0, stream>>>(ids, emb, posmean, pooled);

    pc_kernel<<<N_ / ROWS_, WAVES_ * 32, kPcLds, stream>>>(
        pooled, bWI, bWP, bWU, bW1, initB, predB, updB, s1B, s2W, s2B, out);
}
